// TemporalAttention_16260746183243
// MI455X (gfx1250) — compile-verified, matrix operand order checked by reading, not run on hardware
//
#include <hip/hip_runtime.h>
#include <hip/hip_bf16.h>
#include <stdint.h>

// ---------------- problem constants ----------------
#define BB 2
#define TT 32
#define SS 1024
#define DD 512
#define HH 8
#define HD 64
#define INNER 512        // HH*HD
#define QKVN 1536        // 3*INNER
#define MROWS 65536      // BB*TT*SS

typedef __attribute__((ext_vector_type(16))) __bf16 v16bf;
typedef __attribute__((ext_vector_type(8)))  float  v8f;
typedef __attribute__((ext_vector_type(4)))  unsigned v4u;
typedef __attribute__((ext_vector_type(8)))  int    v8i;
typedef __attribute__((ext_vector_type(4)))  int    v4i;

#if defined(__HIP_DEVICE_COMPILE__) && __has_builtin(__builtin_amdgcn_tensor_load_to_lds)
#define HAS_TDM 1
#else
#define HAS_TDM 0
#endif

// ---------------- helpers ----------------
__device__ __forceinline__ unsigned short f2bf(float f) {
  unsigned u = __builtin_bit_cast(unsigned, f);
  unsigned r = (u + 0x7FFFu + ((u >> 16) & 1u)) >> 16;   // round-to-nearest-even
  return (unsigned short)r;
}
__device__ __forceinline__ unsigned short to_bf16(float f)          { return f2bf(f); }
__device__ __forceinline__ unsigned short to_bf16(unsigned short s) { return s; }

__device__ __forceinline__ void store_c(float* p, float v)          { *p = v; }
__device__ __forceinline__ void store_c(unsigned short* p, float v) { *p = f2bf(v); }

template<typename T> struct is_bf16              { static constexpr bool value = false; };
template<>           struct is_bf16<unsigned short> { static constexpr bool value = true; };

__device__ __forceinline__ unsigned rfl(unsigned x) {
  return (unsigned)__builtin_amdgcn_readfirstlane((int)x);
}

// ---- Tensor Data Mover: 2D bf16 tile (tile_k elems/row x tile_rows rows) from
// global (row stride row_stride_elems) into LDS with post-row padding of 1 DWORD.
// pad_code: (row bytes >> 3) == (1 << pad_code). Resulting LDS row stride:
// tile_k + 2 ushorts (e.g. 32->34, 64->66).  D# layout per CDNA5 ISA ch.8.
__device__ __forceinline__ void tdm_load_2d(void* lds_dst, const void* gsrc,
                                            unsigned tile_k, unsigned tile_rows,
                                            unsigned row_stride_elems,
                                            unsigned pad_code) {
#if HAS_TDM
  uint64_t ga = (uint64_t)(uintptr_t)gsrc;
  v4u g0;
  g0[0] = 1u;                                              // count=1, user desc
  g0[1] = rfl((unsigned)(uintptr_t)lds_dst);               // lds_addr (bytes)
  g0[2] = rfl((unsigned)ga);                               // global_addr[31:0]
  g0[3] = (rfl((unsigned)(ga >> 32)) & 0x1ffffffu) | (2u << 30); // [56:32] | type=2
  v8i g1;
  g1[0] = (int)((1u << 16) | (1u << 20) | (pad_code << 22)); // data_size=2B, pad_en,
                                                             // pad_interval, pad_amount=1DW
  g1[1] = (int)(tile_k << 16);        // tensor_dim0[15:0]
  g1[2] = (int)(tile_rows << 16);     // tensor_dim0 hi=0 | tensor_dim1[15:0]
  g1[3] = (int)(tile_k << 16);        // tensor_dim1 hi=0 | tile_dim0
  g1[4] = (int)tile_rows;             // tile_dim1 | tile_dim2=0
  g1[5] = (int)rfl(row_stride_elems); // tensor_dim0_stride[31:0]
  g1[6] = 0;                          // stride hi | tensor_dim1_stride lo
  g1[7] = 0;
  v4i z4 = {0, 0, 0, 0};
  v8i z8 = {0, 0, 0, 0, 0, 0, 0, 0};
  __builtin_amdgcn_tensor_load_to_lds(g0, g1, z4, z4, z8, 0);
#else
  (void)lds_dst; (void)gsrc; (void)tile_k; (void)tile_rows;
  (void)row_stride_elems; (void)pad_code;
#endif
}

__device__ __forceinline__ void tdm_wait() {
#if HAS_TDM
  __builtin_amdgcn_s_wait_tensorcnt(0);
  asm volatile("" ::: "memory");
#endif
}

union FragU { v16bf v; unsigned u[8]; };

// Build a 16x32 bf16 A/B-style WMMA fragment from an LDS tile stored as
// [row][stride] ushorts. Layout per CDNA5 ISA 7.12.2 (16-bit A 16x32).
__device__ __forceinline__ v16bf load_frag(const unsigned short* lds, int row0,
                                           int stride, int koff) {
  const int lane = threadIdx.x & 31;
  const int half = lane >> 4;
  const int r    = lane & 15;
  const unsigned short* base = lds + (row0 + r) * stride + koff;
  FragU f;
#pragma unroll
  for (int j = 0; j < 8; ++j) {
    int kb = ((j < 4) ? 0 : 16) + half * 8 + 2 * (j & 3);
    f.u[j] = *(const unsigned*)(base + kb);   // 2 consecutive bf16, 4B aligned
  }
  return f.v;
}

__device__ __forceinline__ v8f wmma_bf16(v16bf a, v16bf b, v8f c) {
  return __builtin_amdgcn_wmma_f32_16x16x32_bf16(false, a, false, b,
                                                 (short)0, c, false, false);
}

// ---------------- weight convert+transpose: W (K x N f32) -> WT (N x K bf16) ----
__global__ void convert_transpose(const float* __restrict__ W,
                                  unsigned short* __restrict__ WT,
                                  int K, int N) {
  int idx = blockIdx.x * blockDim.x + threadIdx.x;   // grid covers K*N exactly
  int k = idx / N, n = idx % N;
  WT[(size_t)n * K + k] = f2bf(W[idx]);
}

// ---------------- generic WMMA GEMM: C(MxN) = A(MxK) * BT(NxK)^T ----------------
// A: f32 or bf16(ushort) row-major, row stride K. BT: bf16 N x K row-major.
// Block tile 128x128, BK=32, 8 waves (4x2), each wave 32x64 = 2x4 WMMA tiles.
// Double-buffered LDS; B (and bf16 A) tiles staged by the Tensor Data Mover.
template<typename AT, typename OT>
__global__ __launch_bounds__(256) void gemm_wmma(const AT* __restrict__ A,
                                                 const unsigned short* __restrict__ BT,
                                                 OT* __restrict__ C,
                                                 int M, int N, int K) {
  __shared__ __align__(16) unsigned short sA[2][128 * 34];  // padded stride 34
  __shared__ __align__(16) unsigned short sB[2][128 * 34];
  const int tid  = threadIdx.x;
  const int lane = tid & 31;
  const int wid  = tid >> 5;
  const int wr   = wid >> 1;          // 0..3 : 32-row group
  const int wc   = wid & 1;           // 0..1 : 64-col group
  const int bm   = blockIdx.y * 128;
  const int bn   = blockIdx.x * 128;
  constexpr bool A_IS_BF16 = is_bf16<AT>::value;

  const v8f zero = {0,0,0,0,0,0,0,0};
  v8f acc[2][4];
#pragma unroll
  for (int i = 0; i < 2; ++i)
#pragma unroll
    for (int j = 0; j < 4; ++j) acc[i][j] = zero;

  const int srow = tid >> 1;          // 0..127
  const int scol = (tid & 1) * 16;    // 0 or 16

  auto stage = [&](int k0, int buf) {
#if HAS_TDM
    if (wid == 0) {
      tdm_load_2d(&sB[buf][0], BT + (size_t)bn * K + k0, 32, 128, (unsigned)K, 3);
      if constexpr (A_IS_BF16)
        tdm_load_2d(&sA[buf][0], A + (size_t)bm * K + k0, 32, 128, (unsigned)K, 3);
    }
    if constexpr (!A_IS_BF16) {
      const AT* asrc = A + (size_t)(bm + srow) * K + k0 + scol;
      unsigned short* adst = &sA[buf][srow * 34 + scol];
#pragma unroll
      for (int i = 0; i < 8; ++i) {
        unsigned lo = to_bf16(asrc[2 * i]);
        unsigned hi = to_bf16(asrc[2 * i + 1]);
        *(unsigned*)(adst + 2 * i) = (lo & 0xffffu) | (hi << 16);
      }
    }
#else
    {
      const unsigned short* bsrc = BT + (size_t)(bn + srow) * K + k0 + scol;
      unsigned short* bdst = &sB[buf][srow * 34 + scol];
#pragma unroll
      for (int i = 0; i < 8; ++i)
        *(unsigned*)(bdst + 2 * i) = ((const unsigned*)bsrc)[i];
    }
    {
      const AT* asrc = A + (size_t)(bm + srow) * K + k0 + scol;
      unsigned short* adst = &sA[buf][srow * 34 + scol];
#pragma unroll
      for (int i = 0; i < 8; ++i) {
        unsigned lo = to_bf16(asrc[2 * i]);
        unsigned hi = to_bf16(asrc[2 * i + 1]);
        *(unsigned*)(adst + 2 * i) = (lo & 0xffffu) | (hi << 16);
      }
    }
#endif
  };

  auto compute = [&](int buf) {
    v16bf bfrag[4];
#pragma unroll
    for (int nt = 0; nt < 4; ++nt)
      bfrag[nt] = load_frag(&sB[buf][0], wc * 64 + nt * 16, 34, 0);
#pragma unroll
    for (int mt = 0; mt < 2; ++mt) {
      v16bf a = load_frag(&sA[buf][0], wr * 32 + mt * 16, 34, 0);
#pragma unroll
      for (int nt = 0; nt < 4; ++nt)
        acc[mt][nt] = wmma_bf16(a, bfrag[nt], acc[mt][nt]);
    }
  };

  // software pipeline: TDM/global loads for tile k+1 overlap WMMA on tile k
  stage(0, 0);
#if HAS_TDM
  if (wid == 0) tdm_wait();
#endif
  __syncthreads();
  int buf = 0;
  for (int k0 = 0; k0 < K; k0 += 32) {
    const int nxt = buf ^ 1;
    if (k0 + 32 < K) stage(k0 + 32, nxt);
    compute(buf);
#if HAS_TDM
    if (wid == 0) tdm_wait();
#endif
    __syncthreads();
    buf = nxt;
  }

  // D layout: lanes 0-15 VGPR r -> M=r ; lanes 16-31 VGPR r -> M=8+r ; N=lane&15
  const int half = lane >> 4, nlo = lane & 15;
#pragma unroll
  for (int mt = 0; mt < 2; ++mt)
#pragma unroll
    for (int nt = 0; nt < 4; ++nt)
#pragma unroll
      for (int vr = 0; vr < 8; ++vr) {
        int m = bm + wr * 32 + mt * 16 + half * 8 + vr;
        int n = bn + wc * 64 + nt * 16 + nlo;
        store_c(&C[(size_t)m * N + n], acc[mt][nt][vr]);
      }
}

// ---------------- attention: one wave32 per (b,s,h) head --------------------
// qkv: bf16 [MROWS][1536]; out: bf16 [MROWS][512]
__global__ __launch_bounds__(128) void attn_kernel(const unsigned short* __restrict__ qkv,
                                                   unsigned short* __restrict__ outO) {
  __shared__ __align__(16) unsigned short sQ [4][32 * 66];
  __shared__ __align__(16) unsigned short sK [4][32 * 66];
  __shared__ __align__(16) unsigned short sVT[4][64 * 34];
  __shared__ __align__(16) unsigned short sP [4][32 * 34];

  const int lane = threadIdx.x & 31;
  const int wid  = threadIdx.x >> 5;
  const int gw   = blockIdx.x * 4 + wid;     // 0..16383
  const int h    = gw & 7;
  const int s    = (gw >> 3) & 1023;
  const int b    = gw >> 13;

  unsigned short* Q  = sQ[wid];
  unsigned short* Kd = sK[wid];
  unsigned short* VT = sVT[wid];
  unsigned short* P  = sP[wid];

  const size_t rowstride = (size_t)SS * QKVN;                 // t stride (elems)
  const size_t base0 = ((size_t)b * TT * SS + s) * QKVN + h * HD;

#if HAS_TDM
  // Q,K tiles: 64 bf16/row (128B = 32 DW -> pad_code 4), 32 rows, LDS stride 66
  tdm_load_2d(Q,  qkv + base0,         64, 32, (unsigned)rowstride, 4);
  tdm_load_2d(Kd, qkv + base0 + INNER, 64, 32, (unsigned)rowstride, 4);
  // V: staged manually (transposed, d-major, padded 34)
  for (int t = 0; t < TT; ++t) {
    unsigned v2 = *(const unsigned*)(qkv + base0 + (size_t)t * rowstride + 2 * INNER + 2 * lane);
    VT[(2 * lane)     * 34 + t] = (unsigned short)(v2 & 0xffffu);
    VT[(2 * lane + 1) * 34 + t] = (unsigned short)(v2 >> 16);
  }
  tdm_wait();
#else
  for (int t = 0; t < TT; ++t) {
    size_t rb = base0 + (size_t)t * rowstride;
    unsigned q2 = *(const unsigned*)(qkv + rb + 2 * lane);
    unsigned k2 = *(const unsigned*)(qkv + rb + INNER + 2 * lane);
    unsigned v2 = *(const unsigned*)(qkv + rb + 2 * INNER + 2 * lane);
    *(unsigned*)(Q  + t * 66 + 2 * lane) = q2;
    *(unsigned*)(Kd + t * 66 + 2 * lane) = k2;
    VT[(2 * lane)     * 34 + t] = (unsigned short)(v2 & 0xffffu);
    VT[(2 * lane + 1) * 34 + t] = (unsigned short)(v2 >> 16);
  }
#endif

  // S = Q * K^T  (32x32, K-dim 64) -> 2x2 tiles, 2 k-steps
  const v8f zero = {0,0,0,0,0,0,0,0};
  v8f sacc[2][2] = {{zero, zero}, {zero, zero}};
#pragma unroll
  for (int kd = 0; kd < 64; kd += 32) {
    v16bf a0 = load_frag(Q, 0, 66, kd);
    v16bf a1 = load_frag(Q, 16, 66, kd);
    v16bf b0 = load_frag(Kd, 0, 66, kd);
    v16bf b1 = load_frag(Kd, 16, 66, kd);
    sacc[0][0] = wmma_bf16(a0, b0, sacc[0][0]);
    sacc[0][1] = wmma_bf16(a0, b1, sacc[0][1]);
    sacc[1][0] = wmma_bf16(a1, b0, sacc[1][0]);
    sacc[1][1] = wmma_bf16(a1, b1, sacc[1][1]);
  }

  // causal mask + softmax over the 32 keys of each row; row lives across the
  // 16 lanes of one half-group (shfl_xor 1,2,4,8 stays inside the group)
  const float scale = 0.125f;  // 1/sqrt(64)
  const int half = lane >> 4, nlo = lane & 15;
#pragma unroll
  for (int mt = 0; mt < 2; ++mt)
#pragma unroll
    for (int vr = 0; vr < 8; ++vr) {
      int m = mt * 16 + half * 8 + vr;
      float v0 = sacc[mt][0][vr] * scale;
      float v1 = sacc[mt][1][vr] * scale;
      if (nlo > m)      v0 = -__builtin_inff();
      if (16 + nlo > m) v1 = -__builtin_inff();
      float mx = fmaxf(v0, v1);
#pragma unroll
      for (int off = 1; off < 16; off <<= 1) mx = fmaxf(mx, __shfl_xor(mx, off, 32));
      float e0 = __expf(v0 - mx);
      float e1 = __expf(v1 - mx);
      float sm = e0 + e1;
#pragma unroll
      for (int off = 1; off < 16; off <<= 1) sm += __shfl_xor(sm, off, 32);
      float inv = 1.0f / sm;
      P[m * 34 + nlo]      = f2bf(e0 * inv);
      P[m * 34 + 16 + nlo] = f2bf(e1 * inv);
    }

  // O = P * V  (32x64, K-dim 32) -> 2x4 tiles, single k-step
  v8f oacc[2][4];
#pragma unroll
  for (int i = 0; i < 2; ++i)
#pragma unroll
    for (int j = 0; j < 4; ++j) oacc[i][j] = zero;
  v16bf pa0 = load_frag(P, 0, 34, 0);
  v16bf pa1 = load_frag(P, 16, 34, 0);
#pragma unroll
  for (int nt = 0; nt < 4; ++nt) {
    v16bf vb = load_frag(VT, nt * 16, 34, 0);
    oacc[0][nt] = wmma_bf16(pa0, vb, oacc[0][nt]);
    oacc[1][nt] = wmma_bf16(pa1, vb, oacc[1][nt]);
  }

  // store out_attn[(b,t,s), h*64 + d] bf16
  const size_t obase = ((size_t)b * TT * SS + s) * INNER + h * HD;
  const size_t orow  = (size_t)SS * INNER;
#pragma unroll
  for (int mt = 0; mt < 2; ++mt)
#pragma unroll
    for (int vr = 0; vr < 8; ++vr) {
      int t = mt * 16 + half * 8 + vr;
#pragma unroll
      for (int nt = 0; nt < 4; ++nt)
        outO[obase + (size_t)t * orow + nt * 16 + nlo] = f2bf(oacc[mt][nt][vr]);
    }
}

// ---------------- launcher ----------------
extern "C" void kernel_launch(void* const* d_in, const int* in_sizes, int n_in,
                              void* d_out, int out_size, void* d_ws, size_t ws_size,
                              hipStream_t stream) {
  const float* x     = (const float*)d_in[0];   // (2,32,1024,512) f32
  const float* Wqkv  = (const float*)d_in[1];   // (512,1536) f32
  const float* Wout  = (const float*)d_in[2];   // (512,512) f32
  float* out = (float*)d_out;                   // (2,32,1024,512) f32

  // workspace layout (bf16 intermediates)
  unsigned short* WqkvT = (unsigned short*)d_ws;                    // 1536*512
  unsigned short* WoutT = WqkvT + (size_t)QKVN * DD;                // 512*512
  unsigned short* qkv   = WoutT + (size_t)INNER * DD;               // 65536*1536
  unsigned short* attnO = qkv + (size_t)MROWS * QKVN;               // 65536*512

  convert_transpose<<<(DD * QKVN) / 256, 256, 0, stream>>>(Wqkv, WqkvT, DD, QKVN);
  convert_transpose<<<(INNER * DD) / 256, 256, 0, stream>>>(Wout, WoutT, INNER, DD);

  // qkv = x @ W_qkv   (M=65536, N=1536, K=512), bf16 out
  gemm_wmma<float, unsigned short>
      <<<dim3(QKVN / 128, MROWS / 128), 256, 0, stream>>>(x, WqkvT, qkv,
                                                          MROWS, QKVN, DD);

  // per-(b,s,h) causal attention over T=32
  attn_kernel<<<(BB * SS * HH) / 4, 128, 0, stream>>>(qkv, attnO);

  // out = attn_out @ W_out  (M=65536, N=512, K=512), f32 out
  gemm_wmma<unsigned short, float>
      <<<dim3(DD / 128, MROWS / 128), 256, 0, stream>>>(attnO, WoutT, out,
                                                        MROWS, DD, INNER);
}
